// MASAEmbed_429
// MI455X (gfx1250) — matrix operand order repaired and checked by reading, not compiled, not run on hardware
//
#include <hip/hip_runtime.h>
#include <hip/hip_bf16.h>

typedef __attribute__((ext_vector_type(16))) _Float16 v16h;
typedef __attribute__((ext_vector_type(8)))  _Float16 h8v;
typedef __attribute__((ext_vector_type(8)))  float    v8f;

#define BN_SC 0.99999499f   /* 1/sqrt(1 + 1e-5): eval-mode BN with running stats (0,1) */

struct BNp { const float* g; const float* b; };
struct GcnP {        // raw f32 params
  const float* gcn_w; const float* gcn_b; BNp bn1;
  const float* tcn_w; const float* tcn_b; BNp bn2;
  const float* res_w; const float* res_b; BNp bnr;
};
struct TcnP {
  const float* w5;  const float* b5;  BNp bn5;
  const float* w3a; const float* b3a; BNp bn3a;
  const float* w3b; const float* b3b; BNp bn3b;
  const float* wm;  const float* bm;  BNp bnm;
  const float* wr;  const float* br;  BNp bnr;
};
struct GcnPk {       // f16-packed weights (K zero-padded to mult of 32)
  const _Float16* gcn_w; const float* gcn_b; BNp bn1;
  const _Float16* tcn_w; const float* tcn_b; BNp bn2;
  const _Float16* res_w; const float* res_b; BNp bnr;
};
struct TcnPk {
  const _Float16* w5;  const float* b5;  BNp bn5;
  const _Float16* w3a; const float* b3a; BNp bn3a;
  const _Float16* w3b; const float* b3b; BNp bn3b;
  const _Float16* wm;  const float* bm;  BNp bnm;
  const _Float16* wr;  const float* br;  BNp bnr;
};

__constant__ int HAND_EDGES_C[72] = {
  0,1, 1,2, 2,3, 3,4, 0,5, 5,6, 6,7, 7,8, 0,9, 9,10, 10,11, 11,12,
  0,13, 13,14, 14,15, 15,16, 0,17, 17,18, 18,19, 19,20,
  17,13, 13,9, 9,5, 5,1, 18,14, 14,10, 10,6, 6,2,
  19,15, 15,11, 11,7, 7,3, 20,16, 16,12, 12,8, 8,4};
__constant__ int BODY_EDGES_C[18] = {0,1, 1,2, 2,3, 0,4, 4,5, 5,6, 1,4, 2,5, 3,6};
__constant__ int HAND_POFF[7]  = {0,6,9,12,15,18,21};
__constant__ int HAND_PVERT[21] = {0,1,5,9,13,17, 6,7,8, 18,19,20, 10,11,12, 14,15,16, 2,3,4};
__constant__ int BODY_POFF[4]  = {0,3,4,7};
__constant__ int BODY_PVERT[7] = {1,2,3, 0, 4,5,6};

__device__ inline v16h cat8(h8v lo, h8v hi) {
  return __builtin_shufflevector(lo, hi, 0,1,2,3,4,5,6,7,8,9,10,11,12,13,14,15);
}

// ---- WMMA fragment helpers (ISA 7.12.2 layouts, wave32) ----------------------
// A (16x32 f16), rows in a buffer with row stride `ld` halves:
//   lane m=l&15, h=l>>4; a[0..7]=K 8h+0..7, a[8..15]=K 16+8h+0..7  -> 2x b128
__device__ inline v16h a_frag(const _Float16* buf, int ld, int r0, int k0) {
  int lane = threadIdx.x & 31;
  int m = lane & 15, h = lane >> 4;
  const _Float16* p = buf + (size_t)(r0 + m) * ld + k0 + 8 * h;
  h8v lo = *(const h8v*)p;
  h8v hi = *(const h8v*)(p + 16);
  return cat8(lo, hi);
}
// B (32x16 f16) from packed weights W[o][k] (f16, K mult of 32):
//   lane n=l&15, h=l>>4; b[0..7]=K 8h+0..7, b[8..15]=K 16+8h+0..7  -> 2x b128
__device__ inline v16h b_frag_row(const _Float16* W, int ldk, int n0, int k0) {
  int lane = threadIdx.x & 31;
  int n = lane & 15, hh = lane >> 4;
  const _Float16* p = W + (size_t)(n0 + n) * ldk + k0 + 8 * hh;
  h8v lo = *(const h8v*)p;
  h8v hi = *(const h8v*)(p + 16);
  return cat8(lo, hi);
}

// ---- f32 -> f16 weight packing with K zero-pad ------------------------------
__global__ void pack_w_f16_kernel(const float* src, _Float16* dst, int O, int Ksrc, int Kdst) {
  int idx = blockIdx.x * blockDim.x + threadIdx.x;
  if (idx >= O * Kdst) return;
  int o = idx / Kdst, k = idx % Kdst;
  dst[idx] = (k < Ksrc) ? (_Float16)src[(size_t)o * Ksrc + k] : (_Float16)0.0f;
}

// ---- fused GCN block: graph-matmul (VALU) + 3 channel GEMMs (WMMA) ----------
template<int V>
__device__ void gcn_block_dev(_Float16* pX, _Float16* pH, _Float16* pG,
                              const float* adjS, const GcnPk& gp, int Cin, int Cout) {
  const int ROWS = 16 * V;
  const int Kin = (Cin < 32) ? 32 : Cin;
  const int tid = threadIdx.x;
  const int nthr = blockDim.x;
  // 1) h2 = x ×_v A  -> pH (zero-padded to Kin columns)
  for (int idx = tid; idx < ROWS * Kin; idx += nthr) {
    int r = idx / Kin, c = idx % Kin;
    float s = 0.0f;
    if (c < Cin) {
      int f = r / V, v = r % V;
      const _Float16* base = pX + (f * V) * 128 + c;
      const float* arow = adjS + v * V;
      for (int w = 0; w < V; ++w) s += arow[w] * (float)base[w * 128];
    }
    pH[r * 128 + c] = (_Float16)s;
  }
  __syncthreads();
  const int wv = tid >> 5, nwv = nthr >> 5;
  const int lane = tid & 31;
  const int NT = Cout >> 4, MT = ROWS >> 4;
  // 2) gcn 1x1 GEMM + bias + BN1 + ReLU -> pG
  for (int job = wv; job < MT * NT; job += nwv) {
    int mt = job / NT, nt = job % NT;
    v8f acc = {};
    for (int k0 = 0; k0 < Kin; k0 += 32) {
      v16h a = a_frag(pH, 128, mt * 16, k0);
      v16h b = b_frag_row(gp.gcn_w, Kin, nt * 16, k0);
      acc = __builtin_amdgcn_wmma_f32_16x16x32_f16(false, a, false, b, (short)0, acc, false, false);
    }
    int n = lane & 15, hb = lane >> 4;
    int o = nt * 16 + n;
    float sc = gp.bn1.g[o] * BN_SC, bb = gp.bn1.b[o], bs = gp.gcn_b[o];
#pragma unroll
    for (int r = 0; r < 8; ++r) {
      float y = fmaxf((acc[r] + bs) * sc + bb, 0.0f);
      pG[(mt * 16 + r + 8 * hb) * 128 + o] = (_Float16)y;
    }
  }
  __syncthreads();
  // 3) tcn 1x1 GEMM + BN2  (+)  residual 1x1 GEMM + BNr, ReLU -> pH
  for (int job = wv; job < MT * NT; job += nwv) {
    int mt = job / NT, nt = job % NT;
    v8f a1 = {}, a2 = {};
    for (int k0 = 0; k0 < Cout; k0 += 32) {
      v16h a = a_frag(pG, 128, mt * 16, k0);
      v16h b = b_frag_row(gp.tcn_w, Cout, nt * 16, k0);
      a1 = __builtin_amdgcn_wmma_f32_16x16x32_f16(false, a, false, b, (short)0, a1, false, false);
    }
    for (int k0 = 0; k0 < Kin; k0 += 32) {
      v16h a = a_frag(pX, 128, mt * 16, k0);
      v16h b = b_frag_row(gp.res_w, Kin, nt * 16, k0);
      a2 = __builtin_amdgcn_wmma_f32_16x16x32_f16(false, a, false, b, (short)0, a2, false, false);
    }
    int n = lane & 15, hb = lane >> 4;
    int o = nt * 16 + n;
    float s2 = gp.bn2.g[o] * BN_SC, b2 = gp.bn2.b[o], btc = gp.tcn_b[o];
    float sr = gp.bnr.g[o] * BN_SC, br = gp.bnr.b[o], bR = gp.res_b[o];
#pragma unroll
    for (int r = 0; r < 8; ++r) {
      float y = (a1[r] + btc) * s2 + b2 + (a2[r] + bR) * sr + br;
      pH[(mt * 16 + r + 8 * hb) * 128 + o] = (_Float16)fmaxf(y, 0.0f);
    }
  }
  __syncthreads();
}

// ---- full GCN encoder (data_bn + 3 blocks + part pooling), 16 frames/block --
template<int V, int P, int NE>
__global__ void __launch_bounds__(256)
gcn_encoder_kernel(const float* xa, const float* xb, const float* dbn_g, const float* dbn_b,
                   GcnPk g0, GcnPk g1, GcnPk g2, _Float16* outA, _Float16* outB) {
  const int ROWS = 16 * V;
  extern __shared__ char smem_raw[];
  _Float16* bufX = (_Float16*)smem_raw;
  _Float16* bufH = bufX + ROWS * 128;
  _Float16* bufG = bufH + ROWS * 128;
  float* adjS = (float*)(bufG + ROWS * 128);
  float* dis = adjS + V * V;
  const float* xin = (blockIdx.y == 0) ? xa : xb;
  _Float16* pooled_out = (blockIdx.y == 0) ? outA : outB;
  const int tid = threadIdx.x;
  const int frame0 = blockIdx.x * 16;
  const int* edges = (V == 21) ? HAND_EDGES_C : BODY_EDGES_C;
  // build normalized adjacency D^-1/2 (A+I) D^-1/2
  for (int idx = tid; idx < V * V; idx += blockDim.x)
    adjS[idx] = ((idx / V) == (idx % V)) ? 1.0f : 0.0f;
  __syncthreads();
  if (tid < NE) {
    int i = edges[2 * tid], j = edges[2 * tid + 1];
    adjS[i * V + j] = 1.0f; adjS[j * V + i] = 1.0f;
  }
  __syncthreads();
  if (tid < V) {
    float s = 0.0f;
    for (int j = 0; j < V; ++j) s += adjS[tid * V + j];
    dis[tid] = rsqrtf(s);
  }
  __syncthreads();
  for (int idx = tid; idx < V * V; idx += blockDim.x)
    adjS[idx] *= dis[idx / V] * dis[idx % V];
  // data_bn -> bufX rows (frame,vertex) x 32 (channels 2..31 zero)
  for (int idx = tid; idx < ROWS * 32; idx += blockDim.x) {
    int r = idx >> 5, c = idx & 31;
    float val = 0.0f;
    if (c < 2) {
      int f = r / V, v = r % V;
      int n = frame0 + f;
      float x = xin[(size_t)n * (V * 2) + v * 2 + c];
      val = x * (dbn_g[v * 2 + c] * BN_SC) + dbn_b[v * 2 + c];
    }
    bufX[r * 128 + c] = (_Float16)val;
  }
  __syncthreads();
  _Float16 *pX = bufX, *pH = bufH, *pG = bufG;
  gcn_block_dev<V>(pX, pH, pG, adjS, g0, 2, 32);
  { _Float16* t = pX; pX = pH; pH = t; }
  gcn_block_dev<V>(pX, pH, pG, adjS, g1, 32, 64);
  { _Float16* t = pX; pX = pH; pH = t; }
  gcn_block_dev<V>(pX, pH, pG, adjS, g2, 64, 128);
  { _Float16* t = pX; pX = pH; pH = t; }
  // part-mean pooling -> layout [B][128*P][T] f16 (TCN input, channel = c*P+p)
  const int* poff  = (V == 21) ? HAND_POFF  : BODY_POFF;
  const int* pvert = (V == 21) ? HAND_PVERT : BODY_PVERT;
  for (int idx = tid; idx < 16 * 128 * P; idx += blockDim.x) {
    int f = idx / (128 * P);
    int rem = idx % (128 * P);
    int c = rem / P, p = rem % P;
    int s0 = poff[p], s1 = poff[p + 1];
    float acc = 0.0f;
    for (int q = s0; q < s1; ++q) acc += (float)pX[(f * V + pvert[q]) * 128 + c];
    acc *= 1.0f / (float)(s1 - s0);
    int n = frame0 + f;
    int b = n >> 9, t = n & 511;
    pooled_out[((size_t)b * (128 * P) + (c * P + p)) * 512 + t] = (_Float16)acc;
  }
}

// ---- temporal conv as implicit GEMM (WMMA), fused bias+BN(+ReLU)(+res) ------
// x layout [B][C][T] f16, packed f16 weights [co][ci*kern].
// mode: 0 = BN only, 1 = BN+ReLU, 2 = relu(relu(BN)+res)
__global__ void __launch_bounds__(128)
conv1d_wmma_kernel(const _Float16* x1, const _Float16* x2, int chan1, int ci, int co,
                   const _Float16* wpk, const float* bias, const float* bn_g, const float* bn_b,
                   const _Float16* res, _Float16* out, int kern, int pad, int mode) {
  extern __shared__ char smem_raw[];
  _Float16* xp = (_Float16*)smem_raw;          // [ci][20] halo patch
  const int K = ci * kern;                     // always a multiple of 32 here
  _Float16* BmatT = xp + (size_t)ci * 20;      // [16][K] im2col, transposed (16B aligned)
  const int T = 512;
  const int tid = threadIdx.x;
  const int t0 = blockIdx.x * 16;
  const int b = blockIdx.z;
  const int width = 16 + kern - 1;
  for (int idx = tid; idx < ci * 20; idx += blockDim.x) {
    int c = idx / 20, j = idx % 20;
    float v = 0.0f;
    if (j < width) {
      int t = t0 + j - pad;
      if (t >= 0 && t < T) {
        const _Float16* src = (c < chan1) ? (x1 + ((size_t)b * chan1 + c) * T)
                                          : (x2 + ((size_t)b * (ci - chan1) + (c - chan1)) * T);
        v = (float)src[t];
      }
    }
    xp[idx] = (_Float16)v;
  }
  __syncthreads();
  // im2col transpose: BmatT[n][c*kern+k] = xp[c][n+k]  (built once, shared by 4 waves)
  for (int idx = tid; idx < K * 16; idx += blockDim.x) {
    int kk = idx >> 4, n = idx & 15;
    int c = kk / kern;
    int k = kk - c * kern;
    BmatT[(size_t)n * K + kk] = xp[c * 20 + n + k];
  }
  __syncthreads();
  const int wv = tid >> 5, lane = tid & 31;
  const int o0 = (blockIdx.y * 4 + wv) * 16;
  const int mrow = lane & 15, hh = lane >> 4;
  const _Float16* wrow = wpk + (size_t)(o0 + mrow) * K + 8 * hh;      // A rows
  const _Float16* brow = BmatT + (size_t)(lane & 15) * K + 8 * hh;   // B cols
  v8f acc = {};
  for (int k0 = 0; k0 < K; k0 += 32) {
    __builtin_prefetch(wrow + k0 + 512, 0, 0);
    v16h a = cat8(*(const h8v*)(wrow + k0), *(const h8v*)(wrow + k0 + 16));
    v16h bf = cat8(*(const h8v*)(brow + k0), *(const h8v*)(brow + k0 + 16));
    acc = __builtin_amdgcn_wmma_f32_16x16x32_f16(false, a, false, bf, (short)0, acc, false, false);
  }
  const int t = t0 + (lane & 15);
#pragma unroll
  for (int r = 0; r < 8; ++r) {
    int o = o0 + r + 8 * hh;
    float y = (acc[r] + bias[o]) * (bn_g[o] * BN_SC) + bn_b[o];
    if (mode >= 1) y = fmaxf(y, 0.0f);
    if (mode == 2) y = fmaxf(y + (float)res[((size_t)b * co + o) * T + t], 0.0f);
    out[((size_t)b * co + o) * T + t] = (_Float16)y;
  }
}

// ---- concat 3 streams + sinusoidal positional encoding ----------------------
__global__ void concat_pe_kernel(const _Float16* yr, const _Float16* yl, const _Float16* yb,
                                 float* out) {
  const int T = 512, D = 1536;
  size_t idx = (size_t)blockIdx.x * blockDim.x + threadIdx.x;
  size_t total = (size_t)16 * T * D;
  if (idx >= total) return;
  int j = (int)(idx % D);
  int t = (int)((idx / D) % T);
  int b = (int)(idx / ((size_t)D * T));
  const _Float16* src; int jj;
  if (j < 512)       { src = yr; jj = j; }
  else if (j < 1024) { src = yl; jj = j - 512; }
  else               { src = yb; jj = j - 1024; }
  float v = (float)src[((size_t)b * 512 + jj) * T + t];
  int i2 = (j >> 1) << 1;
  float div = __expf((float)i2 * (-9.210340371976184f / 1536.0f));
  float ang = (float)t * div;
  v += (j & 1) ? __cosf(ang) : __sinf(ang);
  out[idx] = v;
}

// ---- host-side TCN block: 5 fused conv launches -----------------------------
static void run_tcn(hipStream_t stream, const _Float16* x, int ci, int co, const TcnPk& tp,
                    _Float16* tb1, _Float16* tb2a, _Float16* tb2, _Float16* tres, _Float16* y) {
  dim3 blk(128);
  dim3 grid(32, co / 64, 16);
  auto sh = [](int cin, int kern) { return (size_t)(cin * 20 + cin * kern * 16) * sizeof(_Float16); };
  conv1d_wmma_kernel<<<grid, blk, sh(ci, 5), stream>>>(x, nullptr, ci, ci, co, tp.w5, tp.b5, tp.bn5.g, tp.bn5.b, nullptr, tb1, 5, 2, 1);
  conv1d_wmma_kernel<<<grid, blk, sh(ci, 3), stream>>>(x, nullptr, ci, ci, co, tp.w3a, tp.b3a, tp.bn3a.g, tp.bn3a.b, nullptr, tb2a, 3, 1, 1);
  conv1d_wmma_kernel<<<grid, blk, sh(co, 3), stream>>>(tb2a, nullptr, co, co, co, tp.w3b, tp.b3b, tp.bn3b.g, tp.bn3b.b, nullptr, tb2, 3, 1, 1);
  conv1d_wmma_kernel<<<grid, blk, sh(ci, 1), stream>>>(x, nullptr, ci, ci, co, tp.wr, tp.br, tp.bnr.g, tp.bnr.b, nullptr, tres, 1, 0, 0);
  conv1d_wmma_kernel<<<grid, blk, sh(2 * co, 1), stream>>>(tb1, tb2, co, 2 * co, co, tp.wm, tp.bm, tp.bnm.g, tp.bnm.b, tres, y, 1, 0, 2);
}

extern "C" void kernel_launch(void* const* d_in, const int* in_sizes, int n_in,
                              void* d_out, int out_size, void* d_ws, size_t ws_size,
                              hipStream_t stream) {
  (void)in_sizes; (void)n_in; (void)out_size; (void)ws_size;
  const float* rh   = (const float*)d_in[0];
  const float* lh   = (const float*)d_in[1];
  const float* body = (const float*)d_in[2];
  // ---- walk params leaves in setup_inputs() insertion order ----
  int pi = 3;
  auto F = [&]() { return (const float*)d_in[pi++]; };
  auto readBN = [&](BNp& p) { p.g = F(); p.b = F(); };
  auto readGcn = [&](GcnP& p) {
    p.gcn_w = F(); p.gcn_b = F(); readBN(p.bn1);
    p.tcn_w = F(); p.tcn_b = F(); readBN(p.bn2);
    p.res_w = F(); p.res_b = F(); readBN(p.bnr);
  };
  auto readTcn = [&](TcnP& p) {
    p.w5 = F();  p.b5 = F();  readBN(p.bn5);
    p.w3a = F(); p.b3a = F(); readBN(p.bn3a);
    p.w3b = F(); p.b3b = F(); readBN(p.bn3b);
    p.wm = F();  p.bm = F();  readBN(p.bnm);
    p.wr = F();  p.br = F();  readBN(p.bnr);
  };
  BNp hand_dbn, body_dbn;
  GcnP hg[3], bg[3];
  TcnP htc[2], btc[2];
  readBN(hand_dbn);
  for (int k = 0; k < 3; ++k) readGcn(hg[k]);
  for (int k = 0; k < 2; ++k) readTcn(htc[k]);
  readBN(body_dbn);
  for (int k = 0; k < 3; ++k) readGcn(bg[k]);
  for (int k = 0; k < 2; ++k) readTcn(btc[k]);
  // ---- workspace carve (all f16) ----
  _Float16* ws = (_Float16*)d_ws;
  size_t off = 0;
  auto alloc = [&](size_t n) { _Float16* p = ws + off; off += n; return p; };
  _Float16* xr   = alloc((size_t)16 * 768 * 512);
  _Float16* xl   = alloc((size_t)16 * 768 * 512);
  _Float16* xb   = alloc((size_t)16 * 384 * 512);
  _Float16* y1   = alloc((size_t)16 * 256 * 512);
  _Float16* tb1  = alloc((size_t)16 * 512 * 512);
  _Float16* tb2a = alloc((size_t)16 * 512 * 512);
  _Float16* tb2  = alloc((size_t)16 * 512 * 512);
  _Float16* tres = alloc((size_t)16 * 512 * 512);
  _Float16* y2r  = alloc((size_t)16 * 512 * 512);
  _Float16* y2l  = alloc((size_t)16 * 512 * 512);
  _Float16* y2b  = alloc((size_t)16 * 512 * 512);
  // ---- pack all weights to f16 (GCN weights zero-padded to K mult of 32) ----
  auto packW = [&](const float* src, int O, int Ksrc, int Kdst) -> const _Float16* {
    _Float16* dst = alloc((size_t)O * Kdst);
    int total = O * Kdst;
    pack_w_f16_kernel<<<(total + 255) / 256, 256, 0, stream>>>(src, dst, O, Ksrc, Kdst);
    return dst;
  };
  auto packGcn = [&](const GcnP& p, int Cin, int Cout) -> GcnPk {
    int Kpad = (Cin < 32) ? 32 : Cin;
    GcnPk q;
    q.gcn_w = packW(p.gcn_w, Cout, Cin, Kpad); q.gcn_b = p.gcn_b; q.bn1 = p.bn1;
    q.tcn_w = packW(p.tcn_w, Cout, Cout, Cout); q.tcn_b = p.tcn_b; q.bn2 = p.bn2;
    q.res_w = packW(p.res_w, Cout, Cin, Kpad); q.res_b = p.res_b; q.bnr = p.bnr;
    return q;
  };
  auto packTcn = [&](const TcnP& p, int ci, int co) -> TcnPk {
    TcnPk q;
    q.w5  = packW(p.w5,  co, ci * 5, ci * 5); q.b5 = p.b5;  q.bn5 = p.bn5;
    q.w3a = packW(p.w3a, co, ci * 3, ci * 3); q.b3a = p.b3a; q.bn3a = p.bn3a;
    q.w3b = packW(p.w3b, co, co * 3, co * 3); q.b3b = p.b3b; q.bn3b = p.bn3b;
    q.wm  = packW(p.wm,  co, 2 * co, 2 * co); q.bm = p.bm;  q.bnm = p.bnm;
    q.wr  = packW(p.wr,  co, ci, ci);         q.br = p.br;  q.bnr = p.bnr;
    return q;
  };
  const int cin_g[3] = {2, 32, 64}, cout_g[3] = {32, 64, 128};
  GcnPk hgk[3], bgk[3];
  for (int k = 0; k < 3; ++k) hgk[k] = packGcn(hg[k], cin_g[k], cout_g[k]);
  for (int k = 0; k < 3; ++k) bgk[k] = packGcn(bg[k], cin_g[k], cout_g[k]);
  TcnPk htk[2] = {packTcn(htc[0], 768, 256), packTcn(htc[1], 256, 512)};
  TcnPk btk[2] = {packTcn(btc[0], 384, 256), packTcn(btc[1], 256, 512)};
  // ---- stage 1: fused GCN encoders (LDS-resident, WMMA channel GEMMs) ----
  size_t sh_hand = (size_t)3 * 336 * 128 * sizeof(_Float16) + (21 * 21 + 21) * sizeof(float);
  size_t sh_body = (size_t)3 * 112 * 128 * sizeof(_Float16) + (7 * 7 + 7) * sizeof(float);
  gcn_encoder_kernel<21, 6, 36><<<dim3(512, 2), 256, sh_hand, stream>>>(
      rh, lh, hand_dbn.g, hand_dbn.b, hgk[0], hgk[1], hgk[2], xr, xl);
  gcn_encoder_kernel<7, 3, 9><<<dim3(512, 1), 256, sh_body, stream>>>(
      body, body, body_dbn.g, body_dbn.b, bgk[0], bgk[1], bgk[2], xb, xb);
  // ---- stage 2: TCN stacks (implicit-GEMM WMMA convs), temps reused ----
  run_tcn(stream, xr, 768, 256, htk[0], tb1, tb2a, tb2, tres, y1);
  run_tcn(stream, y1, 256, 512, htk[1], tb1, tb2a, tb2, tres, y2r);
  run_tcn(stream, xl, 768, 256, htk[0], tb1, tb2a, tb2, tres, y1);
  run_tcn(stream, y1, 256, 512, htk[1], tb1, tb2a, tb2, tres, y2l);
  run_tcn(stream, xb, 384, 256, btk[0], tb1, tb2a, tb2, tres, y1);
  run_tcn(stream, y1, 256, 512, btk[1], tb1, tb2a, tb2, tres, y2b);
  // ---- stage 3: concat + positional encoding -> f32 output ----
  size_t total = (size_t)16 * 512 * 1536;
  concat_pe_kernel<<<(unsigned)((total + 255) / 256), 256, 0, stream>>>(
      y2r, y2l, y2b, (float*)d_out);
}
